// Action_67362267071068
// MI455X (gfx1250) — matrix operand order repaired and checked by reading, not compiled, not run on hardware
//
#include <hip/hip_runtime.h>
#include <hip/hip_bf16.h>
#include <cmath>

typedef __attribute__((ext_vector_type(16))) _Float16 v16h;
typedef __attribute__((ext_vector_type(8)))  float    v8f;

#define BN_  8      // batch
#define NMOD 5
#define NT   16
#define NHW  49
#define SP   64     // padded spatial
#define NC   256
#define CH   128
#define NPAIR 25

// ---------------------------------------------------------------------------
// WMMA fragment helpers (CDNA5 gfx1250, wave32, 16x16x32 f16 -> f32)
// A: row-major MxK.  16-bit A layout: lane&15 = M row; lane>>4 selects K-half;
//    v16h elem i -> K = (i&7) + ((i>>3)<<4) + 8*(lane>>4)   (two contiguous 8-chunks)
// B: accessed via "Wt" = B transposed (N-major, ld=K): lane&15 = N col;
//    v16h elem i -> K = (i&7) + ((i>>3)<<4) + 8*(lane>>4)   (same as A)
// C/D 16x16 f32: row = r + 8*(lane>>4), col = lane&15
// ---------------------------------------------------------------------------
__device__ __forceinline__ v16h frag_a_h(const _Float16* A, int ld, int row0, int k0, int lane) {
  int m = row0 + (lane & 15);
  int half = lane >> 4;
  const _Float16* p = A + (size_t)m * ld + k0 + half * 8;
  v16h a;
#pragma unroll
  for (int i = 0; i < 8; ++i) a[i] = p[i];
#pragma unroll
  for (int i = 0; i < 8; ++i) a[i + 8] = p[i + 16];
  return a;
}

__device__ __forceinline__ v16h frag_a_f32(const float* A, int ld, int row0, int k0, int lane) {
  int m = row0 + (lane & 15);
  int half = lane >> 4;
  const float* p = A + (size_t)m * ld + k0 + half * 8;
  v16h a;
#pragma unroll
  for (int i = 0; i < 8; ++i) a[i] = (_Float16)p[i];
#pragma unroll
  for (int i = 0; i < 8; ++i) a[i + 8] = (_Float16)p[i + 16];
  return a;
}

__device__ __forceinline__ v16h frag_b_h(const _Float16* Bt, int ld, int n0, int k0, int lane) {
  int n = n0 + (lane & 15);
  int half = lane >> 4;
  const _Float16* p = Bt + (size_t)n * ld + k0 + half * 8;
  v16h b;
#pragma unroll
  for (int i = 0; i < 8; ++i) b[i] = p[i];
#pragma unroll
  for (int i = 0; i < 8; ++i) b[i + 8] = p[i + 16];
  return b;
}

__device__ __forceinline__ v8f wmma16(v16h a, v16h b, v8f c) {
  return __builtin_amdgcn_wmma_f32_16x16x32_f16(false, a, false, b, (short)0, c, false, false);
}

__device__ __forceinline__ float block_reduce_sum(float v, float* sbuf) {
  int tid = threadIdx.x;
  sbuf[tid] = v;
  __syncthreads();
  for (int s = blockDim.x >> 1; s > 0; s >>= 1) {
    if (tid < s) sbuf[tid] += sbuf[tid + s];
    __syncthreads();
  }
  float r = sbuf[0];
  __syncthreads();
  return r;
}

// ---------------------------------------------------------------------------
// Utility kernels
// ---------------------------------------------------------------------------
__global__ void k_zero(float* p, int n) {
  int i = blockIdx.x * blockDim.x + threadIdx.x;
  if (i < n) p[i] = 0.0f;
}

// W: row-major KxN f32 -> Wt: N-major (Wt[n*K+k]) f16
__global__ void k_wt(const float* __restrict__ W, _Float16* __restrict__ Wt, int K, int N) {
  int i = blockIdx.x * blockDim.x + threadIdx.x;
  if (i >= K * N) return;
  int n = i % N, k = i / N;
  Wt[(size_t)n * K + k] = (_Float16)W[(size_t)k * N + n];
}

// Build xf16 (b,m,t,s,c) f16 with s padded to 64 (zeros), and af16 (b,t,s,c)
__global__ void k_prep_x(const float* __restrict__ af, const float* __restrict__ ef,
                         const float* __restrict__ gf, const float* __restrict__ attf,
                         const float* __restrict__ sf,
                         _Float16* __restrict__ xf16, _Float16* __restrict__ af16) {
  size_t idx = (size_t)blockIdx.x * blockDim.x + threadIdx.x;
  const size_t total = (size_t)BN_ * NMOD * NT * SP * NC;
  if (idx >= total) return;
  int c = idx % NC;
  size_t r = idx / NC;
  int s = r % SP; r /= SP;
  int t = r % NT; r /= NT;
  int m = r % NMOD;
  int b = (int)(r / NMOD);
  const float* src = (m == 0) ? af : (m == 1) ? ef : (m == 2) ? gf : (m == 3) ? attf : sf;
  float v = 0.0f;
  if (s < NHW) v = src[((size_t)(b * NC + c) * NT + t) * NHW + s];
  xf16[idx] = (_Float16)v;
  if (m == 0) af16[(((size_t)(b * NT + t)) * SP + s) * NC + c] = (_Float16)v;
}

// f_v = mean over hw of prepped modality features -> x master (f32) + f16 copy
__global__ void k_prep_fv(const float* __restrict__ af, const float* __restrict__ ef,
                          const float* __restrict__ gf, const float* __restrict__ attf,
                          const float* __restrict__ sf,
                          float* __restrict__ x, _Float16* __restrict__ xh) {
  int idx = blockIdx.x * blockDim.x + threadIdx.x;  // ((b*5+m)*16+t)*256+c
  if (idx >= BN_ * NMOD * NT * NC) return;
  int c = idx % NC;
  int r = idx / NC;
  int t = r % NT; r /= NT;
  int m = r % NMOD;
  int b = r / NMOD;
  const float* src = (m == 0) ? af : (m == 1) ? ef : (m == 2) ? gf : (m == 3) ? attf : sf;
  const float* p = src + ((size_t)(b * NC + c) * NT + t) * NHW;
  float s = 0.0f;
#pragma unroll 7
  for (int i = 0; i < NHW; ++i) s += p[i];
  s *= (1.0f / NHW);
  x[idx] = s;
  xh[idx] = (_Float16)s;
}

// ---------------------------------------------------------------------------
// Generic WMMA GEMM: Out[MxN] = A[MxK](f16 row-major) @ B (given as Bt[NxK] f16)
// Each wave computes a 16x64 strip (4 N-tiles) reusing one A fragment per
// K-step (1 A-load + 4 B-loads + 4 WMMA).  N must be a multiple of 64.
// mode 0: f32 row-major; mode 1: f16 row-major; mode 2: f16 transposed in
// 64-row groups (out[grp*N*64 + col*64 + row%64]) for attention V operands.
// ---------------------------------------------------------------------------
__global__ __launch_bounds__(256) void k_gemm(const _Float16* __restrict__ A,
                                              const _Float16* __restrict__ Bt,
                                              const float* __restrict__ bias,
                                              float* __restrict__ outF,
                                              _Float16* __restrict__ outH,
                                              int M, int N, int K, int mode) {
  int gw = (int)((blockIdx.x * blockDim.x + threadIdx.x) >> 5);
  int lane = threadIdx.x & 31;
  int stripsN = N >> 6;
  int tm = gw / stripsN, sn = gw % stripsN;
  if (tm >= (M >> 4)) return;
  v8f acc[4] = {{}, {}, {}, {}};
  for (int k0 = 0; k0 < K; k0 += 32) {
    v16h a = frag_a_h(A, K, tm * 16, k0, lane);
#pragma unroll
    for (int q = 0; q < 4; ++q) {
      v16h b = frag_b_h(Bt, K, sn * 64 + q * 16, k0, lane);
      acc[q] = wmma16(a, b, acc[q]);
    }
  }
  int half = lane >> 4;
#pragma unroll
  for (int q = 0; q < 4; ++q) {
    int col = sn * 64 + q * 16 + (lane & 15);
    float bv = bias ? bias[col] : 0.0f;
    if (mode == 2) {
      int grp = (tm * 16) >> 6;
      int mbase = (tm * 16) & 63;
#pragma unroll
      for (int r = 0; r < 8; ++r)
        outH[((size_t)grp * N + col) * 64 + mbase + r + 8 * half] = (_Float16)(acc[q][r] + bv);
    } else if (mode == 1) {
#pragma unroll
      for (int r = 0; r < 8; ++r)
        outH[(size_t)(tm * 16 + r + 8 * half) * N + col] = (_Float16)(acc[q][r] + bv);
    } else {
#pragma unroll
      for (int r = 0; r < 8; ++r)
        outF[(size_t)(tm * 16 + r + 8 * half) * N + col] = acc[q][r] + bv;
    }
  }
}

// ---------------------------------------------------------------------------
// Attention core pieces (8 waves / block)
// ---------------------------------------------------------------------------
// scores: 4x4 tiles; wave w owns row tm = w&3, cols tnb..tnb+1 (A reused x2)
__device__ void attn_scores(const _Float16* Q, const _Float16* K, float (*S)[SP],
                            int wave, int lane) {
  const float scale = 0.0883883476483f;  // 1/sqrt(128)
  int tm = wave & 3;
  int tnb = (wave >> 2) * 2;
  v8f acc[2] = {{}, {}};
  for (int k0 = 0; k0 < CH; k0 += 32) {
    v16h a = frag_a_h(Q, CH, tm * 16, k0, lane);
#pragma unroll
    for (int q = 0; q < 2; ++q) {
      v16h b = frag_b_h(K, CH, (tnb + q) * 16, k0, lane);  // scores = Q @ K^T
      acc[q] = wmma16(a, b, acc[q]);
    }
  }
  int half = lane >> 4;
#pragma unroll
  for (int q = 0; q < 2; ++q) {
    int col = (tnb + q) * 16 + (lane & 15);
#pragma unroll
    for (int r = 0; r < 8; ++r) {
      float v = acc[q][r] * scale;
      if (col >= NHW) v = -1e30f;
      S[tm * 16 + r + 8 * half][col] = v;
    }
  }
}

__device__ void attn_softmax(float (*S)[SP]) {
  int rid = threadIdx.x;
  if (rid < SP) {
    float mx = -1e30f;
    for (int c = 0; c < NHW; ++c) mx = fmaxf(mx, S[rid][c]);
    float sum = 0.0f;
    for (int c = 0; c < NHW; ++c) {
      float e = __expf(S[rid][c] - mx);
      S[rid][c] = e;
      sum += e;
    }
    float inv = 1.0f / sum;
    for (int c = 0; c < NHW; ++c) S[rid][c] *= inv;
    for (int c = NHW; c < SP; ++c) S[rid][c] = 0.0f;
  }
}

// FAM attention: per block one (b,m,t); writes feat1 f16 (64x256 row-major)
__global__ __launch_bounds__(256) void k_attn_fam(const _Float16* __restrict__ q1,
                                                  const _Float16* __restrict__ k1,
                                                  const _Float16* __restrict__ v1t,
                                                  _Float16* __restrict__ feat1) {
  __shared__ float S[SP][SP];
  int blk = blockIdx.x;                 // (b*5+m)*16+t
  int b = blk / (NMOD * NT);
  int t = blk % NT;
  int kv = b * NT + t;
  const _Float16* Q = q1 + (size_t)blk * SP * CH;
  const _Float16* K = k1 + (size_t)kv * SP * CH;
  const _Float16* Vt = v1t + (size_t)kv * NC * SP;   // [c][s]
  _Float16* O = feat1 + (size_t)blk * SP * NC;
  int wave = threadIdx.x >> 5, lane = threadIdx.x & 31;

  attn_scores(Q, K, S, wave, lane);
  __syncthreads();
  attn_softmax(S);
  __syncthreads();
  // P @ V : wave w owns row tm = w&3, 8 N-tiles (A frag from S reused x8)
  {
    int tm = wave & 3;
    int tnb = (wave >> 2) * 8;
    v8f acc[8] = {{}, {}, {}, {}, {}, {}, {}, {}};
    for (int k0 = 0; k0 < SP; k0 += 32) {
      v16h a = frag_a_f32(&S[0][0], SP, tm * 16, k0, lane);
#pragma unroll
      for (int q = 0; q < 8; ++q) {
        v16h bfr = frag_b_h(Vt, SP, (tnb + q) * 16, k0, lane);
        acc[q] = wmma16(a, bfr, acc[q]);
      }
    }
    int half = lane >> 4;
#pragma unroll
    for (int q = 0; q < 8; ++q) {
      int col = (tnb + q) * 16 + (lane & 15);
#pragma unroll
      for (int r = 0; r < 8; ++r)
        O[(size_t)(tm * 16 + r + 8 * half) * NC + col] = (_Float16)acc[q][r];
    }
  }
}

// ARM attention fused with edge_proj GEMM + hw-mean + BN stat accumulation.
// Per block one (b,pair,t).
__global__ __launch_bounds__(256) void k_attn_arm_edge(
    const _Float16* __restrict__ q2, const _Float16* __restrict__ k2,
    const _Float16* __restrict__ v2t, const _Float16* __restrict__ ept,
    const float* __restrict__ epb, float* __restrict__ Ehw,
    float* __restrict__ S1g, float* __restrict__ S2g) {
  __shared__ float S[SP][SP];
  __shared__ _Float16 Oh[SP][NC];
  __shared__ float colsum[NC];
  __shared__ float sbuf[256];
  int blk = blockIdx.x;                 // (b*25+p)*16+t
  int b = blk / (NPAIR * NT);
  int rr = blk % (NPAIR * NT);
  int p = rr / NT, t = rr % NT;
  int i = p / NMOD, j = p % NMOD;
  const _Float16* Q = q2 + (size_t)((b * NMOD + j) * NT + t) * SP * CH;
  const _Float16* K = k2 + (size_t)((b * NMOD + i) * NT + t) * SP * CH;
  const _Float16* Vt = v2t + (size_t)((b * NMOD + i) * NT + t) * NC * SP;
  int wave = threadIdx.x >> 5, lane = threadIdx.x & 31;
  int half = lane >> 4;

  attn_scores(Q, K, S, wave, lane);
  __syncthreads();
  attn_softmax(S);
  __syncthreads();
  // P @ V -> Oh (f16 in LDS); wave owns tm = w&3, 8 N-tiles
  {
    int tm = wave & 3;
    int tnb = (wave >> 2) * 8;
    v8f acc[8] = {{}, {}, {}, {}, {}, {}, {}, {}};
    for (int k0 = 0; k0 < SP; k0 += 32) {
      v16h a = frag_a_f32(&S[0][0], SP, tm * 16, k0, lane);
#pragma unroll
      for (int q = 0; q < 8; ++q) {
        v16h bfr = frag_b_h(Vt, SP, (tnb + q) * 16, k0, lane);
        acc[q] = wmma16(a, bfr, acc[q]);
      }
    }
#pragma unroll
    for (int q = 0; q < 8; ++q) {
      int col = (tnb + q) * 16 + (lane & 15);
#pragma unroll
      for (int r = 0; r < 8; ++r)
        Oh[tm * 16 + r + 8 * half][col] = (_Float16)acc[q][r];
    }
  }
  __syncthreads();
  colsum[threadIdx.x] = 0.0f;
  __syncthreads();
  float s1 = 0.0f, s2 = 0.0f;
  // edge = Oh @ edge_proj + bias; wave owns tm = w&3, 8 N-tiles (A reused x8)
  {
    int tm = wave & 3;
    int tnb = (wave >> 2) * 8;
    v8f acc[8] = {{}, {}, {}, {}, {}, {}, {}, {}};
    for (int k0 = 0; k0 < NC; k0 += 32) {
      v16h a = frag_a_h(&Oh[0][0], NC, tm * 16, k0, lane);
#pragma unroll
      for (int q = 0; q < 8; ++q) {
        v16h bfr = frag_b_h(ept, NC, (tnb + q) * 16, k0, lane);
        acc[q] = wmma16(a, bfr, acc[q]);
      }
    }
#pragma unroll
    for (int q = 0; q < 8; ++q) {
      int col = (tnb + q) * 16 + (lane & 15);
      float bv = epb[col];
      float csum = 0.0f;
#pragma unroll
      for (int r = 0; r < 8; ++r) {
        int srow = tm * 16 + r + 8 * half;
        float v = acc[q][r] + bv;
        if (srow < NHW) { csum += v; s1 += v; s2 += v * v; }
      }
      atomicAdd(&colsum[col], csum);
    }
  }
  __syncthreads();
  Ehw[(size_t)blk * NC + threadIdx.x] = colsum[threadIdx.x] * (1.0f / NHW);
  float t1 = block_reduce_sum(s1, sbuf);
  float t2 = block_reduce_sum(s2, sbuf);
  if (threadIdx.x == 0) { atomicAdd(&S1g[p], t1); atomicAdd(&S2g[p], t2); }
}

// Apply gem BN affine to hw-mean => f_e (edge master f32 + f16)
__global__ void k_bn_gem(const float* __restrict__ Ehw, const float* __restrict__ S1,
                         const float* __restrict__ S2, const float* __restrict__ g,
                         const float* __restrict__ bt, float* __restrict__ edge,
                         _Float16* __restrict__ edgeh) {
  int row = blockIdx.x;                 // (b*25+p)*16+t
  int c = threadIdx.x;
  int p = (row % (NPAIR * NT)) / NT;
  const float cnt = (float)BN_ * NT * NHW * NC;
  float m = S1[p] / cnt;
  float var = S2[p] / cnt - m * m;
  float inv = rsqrtf(var + 1e-5f);
  size_t idx = (size_t)row * NC + c;
  float e = g[p] * (Ehw[idx] - m) * inv + bt[p];
  edge[idx] = e;
  edgeh[idx] = (_Float16)e;
}

// ---------------------------------------------------------------------------
// GNN layer kernels
// ---------------------------------------------------------------------------
__global__ void k_edge_pre(const float* __restrict__ xA, const float* __restrict__ xB,
                           const float* __restrict__ eE, float* __restrict__ edge1,
                           float* __restrict__ S1e, float* __restrict__ S2e) {
  __shared__ float sbuf[256];
  int row = blockIdx.x;                 // (b*25+p)*16+t
  int c = threadIdx.x;
  int b = row / (NPAIR * NT);
  int rr = row % (NPAIR * NT);
  int p = rr / NT, t = rr % NT;
  int i = p / NMOD, j = p % NMOD;
  size_t ri = (size_t)((b * NMOD + i) * NT + t) * NC + c;
  size_t rj = (size_t)((b * NMOD + j) * NT + t) * NC + c;
  float v = xA[ri] + xB[rj] + eE[(size_t)row * NC + c];
  edge1[(size_t)row * NC + c] = v;
  float t1 = block_reduce_sum(v, sbuf);
  float t2 = block_reduce_sum(v * v, sbuf);
  if (c == 0) { atomicAdd(&S1e[p], t1); atomicAdd(&S2e[p], t2); }
}

__global__ void k_edge_upd(const float* __restrict__ edge1, const float* __restrict__ S1e,
                           const float* __restrict__ S2e, const float* __restrict__ g,
                           const float* __restrict__ bt, float* __restrict__ edge,
                           _Float16* __restrict__ edgeh) {
  int row = blockIdx.x;
  int c = threadIdx.x;
  int p = (row % (NPAIR * NT)) / NT;
  const float cnt = (float)BN_ * NT * NC;
  float m = S1e[p] / cnt;
  float var = S2e[p] / cnt - m * m;
  float inv = rsqrtf(var + 1e-5f);
  size_t idx = (size_t)row * NC + c;
  float e1 = g[p] * (edge1[idx] - m) * inv + bt[p];
  if (isnan(e1)) e1 = 0.0f;
  float en = edge[idx] + fmaxf(e1, 0.0f);
  edge[idx] = en;
  edgeh[idx] = (_Float16)en;
}

__global__ void k_agg_node(const float* __restrict__ edge, const float* __restrict__ xV,
                           const float* __restrict__ xU, float* __restrict__ xn,
                           float* __restrict__ S1v, float* __restrict__ S2v) {
  __shared__ float sbuf[256];
  int row = blockIdx.x;                 // (b*5+i)*16+t
  int c = threadIdx.x;
  int b = row / (NMOD * NT);
  int rr = row % (NMOD * NT);
  int i = rr / NT, t = rr % NT;
  float sg[NMOD];
  float esum = 0.0f;
#pragma unroll
  for (int j = 0; j < NMOD; ++j) {
    size_t eidx = (size_t)((b * NPAIR + i * NMOD + j) * NT + t) * NC + c;
    float e = edge[eidx];
    float s = 1.0f / (1.0f + __expf(-e));   // sigmoid in (0,1)
    sg[j] = __expf(s);
    esum += sg[j];
  }
  float agg = 0.0f;
#pragma unroll
  for (int j = 0; j < NMOD; ++j) {
    size_t vj = (size_t)((b * NMOD + j) * NT + t) * NC + c;
    agg += (sg[j] / esum) * xV[vj];
  }
  agg *= (1.0f / NMOD);
  size_t idx = (size_t)row * NC + c;
  float v = xU[idx] + agg;
  xn[idx] = v;
  float t1 = block_reduce_sum(v, sbuf);
  float t2 = block_reduce_sum(v * v, sbuf);
  if (c == 0) { atomicAdd(&S1v[i], t1); atomicAdd(&S2v[i], t2); }
}

__global__ void k_node_upd(const float* __restrict__ xn, const float* __restrict__ S1v,
                           const float* __restrict__ S2v, const float* __restrict__ g,
                           const float* __restrict__ bt, float* __restrict__ x,
                           _Float16* __restrict__ xh, int maskNode) {
  int row = blockIdx.x;
  int c = threadIdx.x;
  int i = (row % (NMOD * NT)) / NT;
  const float cnt = (float)BN_ * NT * NC;
  float m = S1v[i] / cnt;
  float var = S2v[i] / cnt - m * m;
  float inv = rsqrtf(var + 1e-5f);
  size_t idx = (size_t)row * NC + c;
  float x1 = g[i] * (xn[idx] - m) * inv + bt[i];
  if (maskNode && isnan(x1)) x1 = 0.0f;
  float v = fmaxf(x[idx] + x1, 0.0f);
  x[idx] = v;
  xh[idx] = (_Float16)v;
}

// ---------------------------------------------------------------------------
// Output heads: cl (640) then cl_edge (3200)
// ---------------------------------------------------------------------------
__global__ void k_heads(const float* __restrict__ x, const float* __restrict__ sc,
                        const float* __restrict__ edge, const float* __restrict__ efw,
                        const float* __restrict__ efb, float* __restrict__ out) {
  int idx = blockIdx.x * blockDim.x + threadIdx.x;
  if (idx < BN_ * NMOD * NT) {
    const float* xr = x + (size_t)idx * NC;
    int i = (idx / NT) % NMOD;
    const float* s = sc + (size_t)i * NC;
    float xx = 0.0f, ss = 0.0f, xs = 0.0f;
    for (int c = 0; c < NC; ++c) {
      float a = xr[c], b = s[c];
      xx += a * a; ss += b * b; xs += a * b;
    }
    float d = fmaxf(sqrtf(xx), 1e-12f) * fmaxf(sqrtf(ss), 1e-12f);
    out[idx] = xs / d;
  } else if (idx < BN_ * NMOD * NT + BN_ * NPAIR * NT) {
    int e = idx - BN_ * NMOD * NT;
    const float* er = edge + (size_t)e * NC;
    float a = 0.0f;
    for (int c = 0; c < NC; ++c) a += er[c] * efw[c];
    out[idx] = a + efb[0];
  }
}

// ---------------------------------------------------------------------------
// Host launch
// ---------------------------------------------------------------------------
static void launch_gemm(const _Float16* A, const _Float16* Bt, const float* bias,
                        float* outF, _Float16* outH, int M, int N, int K, int mode,
                        hipStream_t stream) {
  int strips = (M >> 4) * (N >> 6);
  int blocks = (strips + 7) / 8;
  k_gemm<<<blocks, 256, 0, stream>>>(A, Bt, bias, outF, outH, M, N, K, mode);
}

extern "C" void kernel_launch(void* const* d_in, const int* in_sizes, int n_in,
                              void* d_out, int out_size, void* d_ws, size_t ws_size,
                              hipStream_t stream) {
  (void)in_sizes; (void)n_in; (void)out_size; (void)ws_size;
  const float* af   = (const float*)d_in[0];
  const float* ef   = (const float*)d_in[1];
  const float* gf   = (const float*)d_in[2];
  const float* attf = (const float*)d_in[3];
  const float* sf   = (const float*)d_in[4];
  const float* epb     = (const float*)d_in[12];
  const float* gem_g   = (const float*)d_in[13];
  const float* gem_b   = (const float*)d_in[14];
  const float* bnv1_g  = (const float*)d_in[25];
  const float* bnv1_b  = (const float*)d_in[26];
  const float* bne1_g  = (const float*)d_in[27];
  const float* bne1_b  = (const float*)d_in[28];
  const float* bnv2_g  = (const float*)d_in[29];
  const float* bnv2_b  = (const float*)d_in[30];
  const float* bne2_g  = (const float*)d_in[31];
  const float* bne2_b  = (const float*)d_in[32];
  const float* sc      = (const float*)d_in[33];
  const float* efw     = (const float*)d_in[34];
  const float* efb     = (const float*)d_in[35];
  float* out = (float*)d_out;

  // ---- workspace bump allocator ----
  char* base = (char*)d_ws;
  size_t off = 0;
  auto alloc = [&](size_t bytes) -> void* {
    void* p = base + off;
    off = (off + bytes + 255) & ~(size_t)255;
    return p;
  };
  auto allocH = [&](size_t n) { return (_Float16*)alloc(n * sizeof(_Float16)); };
  auto allocF = [&](size_t n) { return (float*)alloc(n * sizeof(float)); };

  // transposed f16 weights
  struct WtDesc { int in_idx; int K; int N; _Float16* dst; };
  WtDesc wds[17] = {
      {5, NC, CH, nullptr},  {6, NC, CH, nullptr},  {7, NC, NC, nullptr},
      {8, NC, CH, nullptr},  {9, NC, CH, nullptr},  {10, NC, NC, nullptr},
      {11, NC, NC, nullptr},
      {15, NC, NC, nullptr}, {16, NC, NC, nullptr}, {17, NC, NC, nullptr},
      {18, NC, NC, nullptr}, {19, NC, NC, nullptr},
      {20, NC, NC, nullptr}, {21, NC, NC, nullptr}, {22, NC, NC, nullptr},
      {23, NC, NC, nullptr}, {24, NC, NC, nullptr}};
  for (int w = 0; w < 17; ++w) wds[w].dst = allocH((size_t)wds[w].K * wds[w].N);
  _Float16* fam_q_t = wds[0].dst;  _Float16* fam_k_t = wds[1].dst;
  _Float16* fam_v_t = wds[2].dst;  _Float16* arm_q_t = wds[3].dst;
  _Float16* arm_k_t = wds[4].dst;  _Float16* arm_v_t = wds[5].dst;
  _Float16* ep_t    = wds[6].dst;
  _Float16* U1t = wds[7].dst;  _Float16* V1t = wds[8].dst;  _Float16* A1t = wds[9].dst;
  _Float16* B1t = wds[10].dst; _Float16* E1t = wds[11].dst;
  _Float16* U2t = wds[12].dst; _Float16* V2t = wds[13].dst; _Float16* A2t = wds[14].dst;
  _Float16* B2t = wds[15].dst; _Float16* E2t = wds[16].dst;

  const size_t nBMT  = (size_t)BN_ * NMOD * NT;          // 640
  const size_t nBPT  = (size_t)BN_ * NPAIR * NT;         // 3200
  const size_t nBT   = (size_t)BN_ * NT;                 // 128
  _Float16* xf16  = allocH(nBMT * SP * NC);
  _Float16* af16  = allocH(nBT * SP * NC);
  float*    x     = allocF(nBMT * NC);
  _Float16* xh    = allocH(nBMT * NC);
  _Float16* q1    = allocH(nBMT * SP * CH);
  _Float16* k1    = allocH(nBT * SP * CH);
  _Float16* v1t   = allocH(nBT * NC * SP);
  _Float16* feat1 = allocH(nBMT * SP * NC);
  _Float16* q2    = allocH(nBMT * SP * CH);
  _Float16* k2    = allocH(nBMT * SP * CH);
  _Float16* v2t   = allocH(nBMT * NC * SP);
  float*    Ehw   = allocF(nBPT * NC);
  float*    edge  = allocF(nBPT * NC);
  _Float16* edgeh = allocH(nBPT * NC);
  float*    xA    = allocF(nBMT * NC);
  float*    xB    = allocF(nBMT * NC);
  float*    xV    = allocF(nBMT * NC);
  float*    xU    = allocF(nBMT * NC);
  float*    eE    = allocF(nBPT * NC);
  float*    edge1 = allocF(nBPT * NC);
  float*    xn    = allocF(nBMT * NC);
  float*    stats = allocF(256);
  float* S1g = stats;        float* S2g = stats + 25;
  float* S1e1 = stats + 50;  float* S2e1 = stats + 75;
  float* S1v1 = stats + 100; float* S2v1 = stats + 105;
  float* S1e2 = stats + 110; float* S2e2 = stats + 135;
  float* S1v2 = stats + 160; float* S2v2 = stats + 165;

  // ---- pipeline ----
  k_zero<<<1, 256, 0, stream>>>(stats, 256);
  for (int w = 0; w < 17; ++w) {
    int n = wds[w].K * wds[w].N;
    k_wt<<<(n + 255) / 256, 256, 0, stream>>>((const float*)d_in[wds[w].in_idx],
                                              wds[w].dst, wds[w].K, wds[w].N);
  }
  {
    size_t tot = nBMT * SP * NC;
    k_prep_x<<<(int)((tot + 255) / 256), 256, 0, stream>>>(af, ef, gf, attf, sf, xf16, af16);
    k_prep_fv<<<(int)((nBMT * NC + 255) / 256), 256, 0, stream>>>(af, ef, gf, attf, sf, x, xh);
  }
  // FAM projections + attention
  launch_gemm(xf16, fam_q_t, nullptr, nullptr, q1, (int)(nBMT * SP), CH, NC, 1, stream);
  launch_gemm(af16, fam_k_t, nullptr, nullptr, k1, (int)(nBT * SP), CH, NC, 1, stream);
  launch_gemm(af16, fam_v_t, nullptr, nullptr, v1t, (int)(nBT * SP), NC, NC, 2, stream);
  k_attn_fam<<<(int)nBMT, 256, 0, stream>>>(q1, k1, v1t, feat1);
  // ARM projections + attention + edge_proj + BN stats
  launch_gemm(feat1, arm_q_t, nullptr, nullptr, q2, (int)(nBMT * SP), CH, NC, 1, stream);
  launch_gemm(feat1, arm_k_t, nullptr, nullptr, k2, (int)(nBMT * SP), CH, NC, 1, stream);
  launch_gemm(feat1, arm_v_t, nullptr, nullptr, v2t, (int)(nBMT * SP), NC, NC, 2, stream);
  k_attn_arm_edge<<<(int)nBPT, 256, 0, stream>>>(q2, k2, v2t, ep_t, epb, Ehw, S1g, S2g);
  k_bn_gem<<<(int)nBPT, 256, 0, stream>>>(Ehw, S1g, S2g, gem_g, gem_b, edge, edgeh);

  // GNN layer 1
  launch_gemm(xh, A1t, nullptr, xA, nullptr, (int)nBMT, NC, NC, 0, stream);
  launch_gemm(xh, B1t, nullptr, xB, nullptr, (int)nBMT, NC, NC, 0, stream);
  launch_gemm(xh, V1t, nullptr, xV, nullptr, (int)nBMT, NC, NC, 0, stream);
  launch_gemm(xh, U1t, nullptr, xU, nullptr, (int)nBMT, NC, NC, 0, stream);
  launch_gemm(edgeh, E1t, nullptr, eE, nullptr, (int)nBPT, NC, NC, 0, stream);
  k_edge_pre<<<(int)nBPT, 256, 0, stream>>>(xA, xB, eE, edge1, S1e1, S2e1);
  k_edge_upd<<<(int)nBPT, 256, 0, stream>>>(edge1, S1e1, S2e1, bne1_g, bne1_b, edge, edgeh);
  k_agg_node<<<(int)nBMT, 256, 0, stream>>>(edge, xV, xU, xn, S1v1, S2v1);
  k_node_upd<<<(int)nBMT, 256, 0, stream>>>(xn, S1v1, S2v1, bnv1_g, bnv1_b, x, xh, 1);

  // GNN layer 2
  launch_gemm(xh, A2t, nullptr, xA, nullptr, (int)nBMT, NC, NC, 0, stream);
  launch_gemm(xh, B2t, nullptr, xB, nullptr, (int)nBMT, NC, NC, 0, stream);
  launch_gemm(xh, V2t, nullptr, xV, nullptr, (int)nBMT, NC, NC, 0, stream);
  launch_gemm(xh, U2t, nullptr, xU, nullptr, (int)nBMT, NC, NC, 0, stream);
  launch_gemm(edgeh, E2t, nullptr, eE, nullptr, (int)nBPT, NC, NC, 0, stream);
  k_edge_pre<<<(int)nBPT, 256, 0, stream>>>(xA, xB, eE, edge1, S1e2, S2e2);
  k_edge_upd<<<(int)nBPT, 256, 0, stream>>>(edge1, S1e2, S2e2, bne2_g, bne2_b, edge, edgeh);
  k_agg_node<<<(int)nBMT, 256, 0, stream>>>(edge, xV, xU, xn, S1v2, S2v2);
  k_node_upd<<<(int)nBMT, 256, 0, stream>>>(xn, S1v2, S2v2, bnv2_g, bnv2_b, x, xh, 0);

  // heads
  int totOut = (int)(nBMT + nBPT);
  k_heads<<<(totOut + 255) / 256, 256, 0, stream>>>(x, sc, edge, efw, efb, out);
}
